// CombinedModel_10264971837882
// MI455X (gfx1250) — matrix operand order repaired and checked by reading, not compiled, not run on hardware
//
#include <hip/hip_runtime.h>
#include <math.h>

// ---------------- problem dims ----------------
#define B_SZ     8
#define L_SZ     2048
#define D_MODEL  1024
#define D_INNER  2048
#define NHEADS   32
#define HEADDIM  64
#define D_STATE  128
#define CONV_DIM 2304            // D_INNER + 2*D_STATE
#define D_IN_PROJ 4384           // 2*D_INNER + 2*D_STATE + NHEADS
#define ROWS     (B_SZ * L_SZ)   // 16384
#define EPS      1e-5f

typedef __attribute__((ext_vector_type(16))) __bf16 v16bf;
typedef __attribute__((ext_vector_type(8)))  float  v8f;

union FragBF { v16bf v; unsigned int u[8]; };

// ---------------- bf16 split helpers ----------------
__device__ __forceinline__ unsigned short f32_to_bf16_rne(float f) {
    unsigned int u = __float_as_uint(f);
    u += 0x7FFFu + ((u >> 16) & 1u);   // round-to-nearest-even
    return (unsigned short)(u >> 16);
}
__device__ __forceinline__ float bf16_to_f32(unsigned short h) {
    return __uint_as_float(((unsigned int)h) << 16);
}

// split fp32 -> (hi, lo) bf16 pair so hi+lo ~= fp32
__global__ __launch_bounds__(256)
void split_bf16_kernel(const float* __restrict__ in,
                       unsigned short* __restrict__ hi,
                       unsigned short* __restrict__ lo, int n) {
    int i = blockIdx.x * 256 + threadIdx.x;
    if (i >= n) return;
    float f = in[i];
    unsigned short h = f32_to_bf16_rne(f);
    hi[i] = h;
    lo[i] = f32_to_bf16_rne(f - bf16_to_f32(h));
}

// ---------------- fragment loaders (layouts per CDNA5 ISA 7.12.2) ----------------
// A 16x32 bf16: lane m=lane&15, hi=lane>>4; dwords at K-groups {ko+hi*8+2j} and {ko+16+hi*8+2j}
__device__ __forceinline__ void load_afrag(FragBF& f, const unsigned int* __restrict__ row, int a0) {
#pragma unroll
    for (int j = 0; j < 4; ++j) { f.u[j] = row[a0 + j]; f.u[4 + j] = row[a0 + 8 + j]; }
}
// B 32x16 bf16 (from W row-major, used as W^T): dwords at K-groups {ko+hi*8+2j} and {ko+16+hi*8+2j}
__device__ __forceinline__ void load_bfrag(FragBF& f, const unsigned int* __restrict__ row, int b0) {
#pragma unroll
    for (int j = 0; j < 4; ++j) { f.u[j] = row[b0 + j]; f.u[4 + j] = row[b0 + 8 + j]; }
}
// C/D f32: VGPR r -> M = r + 8*hi, N = lane&15
__device__ __forceinline__ void store_tile(float* __restrict__ C, int N, int m, int n, v8f acc) {
    float* crow = C + (size_t)m * N + n;
#pragma unroll
    for (int r = 0; r < 8; ++r) crow[(size_t)r * N] = acc[r];
}

// ---- WMMA GEMM: C[M,N] = A[M,K] * W[N,K]^T, split-bf16 (3 WMMA per product),
// ---- 32x32 output tile per wave (4 accumulators, 12 WMMA per 16 b128 loads)
__global__ __launch_bounds__(256)
void gemm_bf16x3_kernel(const unsigned short* __restrict__ Ah,
                        const unsigned short* __restrict__ Al,
                        const unsigned short* __restrict__ Bh,
                        const unsigned short* __restrict__ Bl,
                        float* __restrict__ C, int N, int K) {
    const int lane = threadIdx.x & 31;
    const int wave = threadIdx.x >> 5;
    const int m0   = (blockIdx.x * 8 + wave) * 32;   // 32 rows per wave
    const int n0   = blockIdx.y * 32;                // 32 cols per wave
    const int r16  = lane & 15;
    const int hi   = lane >> 4;

    const unsigned int* a0h = (const unsigned int*)(Ah + (size_t)(m0 + r16) * K);
    const unsigned int* a0l = (const unsigned int*)(Al + (size_t)(m0 + r16) * K);
    const unsigned int* a1h = (const unsigned int*)(Ah + (size_t)(m0 + 16 + r16) * K);
    const unsigned int* a1l = (const unsigned int*)(Al + (size_t)(m0 + 16 + r16) * K);
    const unsigned int* b0h = (const unsigned int*)(Bh + (size_t)(n0 + r16) * K);
    const unsigned int* b0l = (const unsigned int*)(Bl + (size_t)(n0 + r16) * K);
    const unsigned int* b1h = (const unsigned int*)(Bh + (size_t)(n0 + 16 + r16) * K);
    const unsigned int* b1l = (const unsigned int*)(Bl + (size_t)(n0 + 16 + r16) * K);

    v8f c00 = {}, c01 = {}, c10 = {}, c11 = {};
    for (int ko = 0; ko < K; ko += 32) {
        const int a0 = (ko >> 1) + hi * 4;
        const int b0 = (ko >> 1) + hi * 4;
        FragBF A0h, A0l, A1h, A1l, B0h, B0l, B1h, B1l;
        load_afrag(A0h, a0h, a0);  load_afrag(A0l, a0l, a0);
        load_afrag(A1h, a1h, a0);  load_afrag(A1l, a1l, a0);
        load_bfrag(B0h, b0h, b0);  load_bfrag(B0l, b0l, b0);
        load_bfrag(B1h, b1h, b0);  load_bfrag(B1l, b1l, b0);
        // A*B ~= Ah*Bh + Ah*Bl + Al*Bh  (fp32-accurate on the bf16 matrix pipes)
        c00 = __builtin_amdgcn_wmma_f32_16x16x32_bf16(false, A0h.v, false, B0h.v, (short)0, c00, false, false);
        c00 = __builtin_amdgcn_wmma_f32_16x16x32_bf16(false, A0h.v, false, B0l.v, (short)0, c00, false, false);
        c00 = __builtin_amdgcn_wmma_f32_16x16x32_bf16(false, A0l.v, false, B0h.v, (short)0, c00, false, false);
        c01 = __builtin_amdgcn_wmma_f32_16x16x32_bf16(false, A0h.v, false, B1h.v, (short)0, c01, false, false);
        c01 = __builtin_amdgcn_wmma_f32_16x16x32_bf16(false, A0h.v, false, B1l.v, (short)0, c01, false, false);
        c01 = __builtin_amdgcn_wmma_f32_16x16x32_bf16(false, A0l.v, false, B1h.v, (short)0, c01, false, false);
        c10 = __builtin_amdgcn_wmma_f32_16x16x32_bf16(false, A1h.v, false, B0h.v, (short)0, c10, false, false);
        c10 = __builtin_amdgcn_wmma_f32_16x16x32_bf16(false, A1h.v, false, B0l.v, (short)0, c10, false, false);
        c10 = __builtin_amdgcn_wmma_f32_16x16x32_bf16(false, A1l.v, false, B0h.v, (short)0, c10, false, false);
        c11 = __builtin_amdgcn_wmma_f32_16x16x32_bf16(false, A1h.v, false, B1h.v, (short)0, c11, false, false);
        c11 = __builtin_amdgcn_wmma_f32_16x16x32_bf16(false, A1h.v, false, B1l.v, (short)0, c11, false, false);
        c11 = __builtin_amdgcn_wmma_f32_16x16x32_bf16(false, A1l.v, false, B1h.v, (short)0, c11, false, false);
    }
    store_tile(C, N, m0 + 8 * hi,      n0 + r16,      c00);
    store_tile(C, N, m0 + 8 * hi,      n0 + 16 + r16, c01);
    store_tile(C, N, m0 + 16 + 8 * hi, n0 + r16,      c10);
    store_tile(C, N, m0 + 16 + 8 * hi, n0 + 16 + r16, c11);
}

// ---------------- dt = softplus(dt_raw + dt_bias) ----------------
__global__ __launch_bounds__(256)
void dt_softplus_kernel(const float* __restrict__ zx, const float* __restrict__ dt_bias,
                        float* __restrict__ dt_out) {
    int i = blockIdx.x * 256 + threadIdx.x;
    if (i >= ROWS * NHEADS) return;
    int row = i >> 5, h = i & 31;
    float v = zx[(size_t)row * D_IN_PROJ + (2 * D_INNER + 2 * D_STATE) + h] + dt_bias[h];
    dt_out[i] = (v > 20.0f) ? v : log1pf(expf(v));
}

// ---------------- depthwise causal conv(k=4) + bias + silu ----------------
__global__ __launch_bounds__(256)
void conv_silu_kernel(const float* __restrict__ zx, const float* __restrict__ w,
                      const float* __restrict__ bias, float* __restrict__ out) {
    int c = blockIdx.x * 256 + threadIdx.x;
    int t = blockIdx.y;                       // global row b*L + l
    if (c >= CONV_DIM) return;
    int l = t & (L_SZ - 1);
    const float* col = zx + D_INNER + c;      // xBC column in zxbcdt
    float acc = bias[c];
#pragma unroll
    for (int k = 0; k < 4; ++k) {
        int ll = l - 3 + k;
        if (ll >= 0) acc += w[c * 4 + k] * col[(size_t)(t - 3 + k) * D_IN_PROJ];
    }
    out[(size_t)t * CONV_DIM + c] = acc / (1.0f + expf(-acc));   // silu
}

// ---------------- sequential SSD scan: one workgroup per (batch, head) ----------------
__global__ __launch_bounds__(256)
void ssd_scan_kernel(const float* __restrict__ xconv, const float* __restrict__ dt,
                     const float* __restrict__ A_log, const float* __restrict__ D_param,
                     float* __restrict__ y) {
    const int bh = blockIdx.x;                // 0..255
    const int b = bh >> 5, h = bh & 31;
    const int tid = threadIdx.x;
    const int p = tid >> 2;                   // headdim index 0..63
    const int q = tid & 3;                    // state-chunk 0..3
    const int n0 = q * 32;
    __shared__ float sB[2][D_STATE], sC[2][D_STATE];

    const float Ah = -expf(A_log[h]);
    const float Dh = D_param[h];
    float hreg[32];
#pragma unroll
    for (int j = 0; j < 32; ++j) hreg[j] = 0.0f;

    for (int t = 0; t < L_SZ; ++t) {
        const size_t row = (size_t)(b * L_SZ + t);
        const float* xrow = xconv + row * CONV_DIM;
        const int buf = t & 1;
        if (tid < 128)      sB[buf][tid]        = xrow[D_INNER + tid];
        else                sC[buf][tid - 128]  = xrow[D_INNER + D_STATE + (tid - 128)];
        float dtv = dt[row * NHEADS + h];
        float dA  = expf(dtv * Ah);
        float xv  = xrow[(size_t)h * HEADDIM + p];
        float dtx = dtv * xv;
        __syncthreads();
        float yp = 0.0f;
#pragma unroll
        for (int j = 0; j < 32; ++j) {
            float Bv = sB[buf][n0 + j];
            float Cv = sC[buf][n0 + j];
            hreg[j] = fmaf(dA, hreg[j], dtx * Bv);
            yp = fmaf(hreg[j], Cv, yp);
        }
        yp += __shfl_xor(yp, 1, 32);
        yp += __shfl_xor(yp, 2, 32);
        if (q == 0) y[row * D_INNER + h * HEADDIM + p] = yp + Dh * xv;
    }
}

// ---------------- y*silu(z) -> RMSNorm(gnorm_w) -> split-bf16 ----------------
__global__ __launch_bounds__(256)
void gate_gnorm_kernel(const float* __restrict__ y, const float* __restrict__ zx,
                       const float* __restrict__ gw,
                       unsigned short* __restrict__ Yh, unsigned short* __restrict__ Yl) {
    const int row = blockIdx.x, tid = threadIdx.x;
    const float* yrow = y  + (size_t)row * D_INNER;
    const float* zrow = zx + (size_t)row * D_IN_PROJ;
    float u[8]; float ss = 0.0f;
#pragma unroll
    for (int k = 0; k < 8; ++k) {
        int i = tid + k * 256;
        float zv = zrow[i];
        float uv = yrow[i] * (zv / (1.0f + expf(-zv)));
        u[k] = uv; ss += uv * uv;
    }
    __shared__ float red[256];
    red[tid] = ss; __syncthreads();
    for (int s = 128; s > 0; s >>= 1) { if (tid < s) red[tid] += red[tid + s]; __syncthreads(); }
    float scale = rsqrtf(red[0] / (float)D_INNER + EPS);
#pragma unroll
    for (int k = 0; k < 8; ++k) {
        int i = tid + k * 256;
        float v = u[k] * scale * gw[i];
        unsigned short hh = f32_to_bf16_rne(v);
        Yh[(size_t)row * D_INNER + i] = hh;
        Yl[(size_t)row * D_INNER + i] = f32_to_bf16_rne(v - bf16_to_f32(hh));
    }
}

// ---------------- RMSNorm(norm_w) in place, fold 1/L for the mean ----------------
__global__ __launch_bounds__(256)
void rmsnorm_rows_kernel(float* __restrict__ om, const float* __restrict__ nw) {
    const int row = blockIdx.x, tid = threadIdx.x;
    float* r = om + (size_t)row * D_MODEL;
    float v[4]; float ss = 0.0f;
#pragma unroll
    for (int k = 0; k < 4; ++k) { v[k] = r[tid + k * 256]; ss += v[k] * v[k]; }
    __shared__ float red[256];
    red[tid] = ss; __syncthreads();
    for (int s = 128; s > 0; s >>= 1) { if (tid < s) red[tid] += red[tid + s]; __syncthreads(); }
    float scale = rsqrtf(red[0] / (float)D_MODEL + EPS) * (1.0f / (float)L_SZ);
#pragma unroll
    for (int k = 0; k < 4; ++k) r[tid + k * 256] = v[k] * scale * nw[tid + k * 256];
}

__global__ __launch_bounds__(256)
void mean_reduce_kernel(const float* __restrict__ om, float* __restrict__ mean) {
    int i = blockIdx.x * 256 + threadIdx.x;
    if (i >= B_SZ * D_MODEL) return;
    int b = i >> 10, d = i & (D_MODEL - 1);
    const float* base = om + (size_t)b * L_SZ * D_MODEL + d;
    float s = 0.0f;
    for (int l = 0; l < L_SZ; ++l) s += base[(size_t)l * D_MODEL];
    mean[i] = s;
}

__global__ void cls_kernel(const float* __restrict__ mean, const float* __restrict__ cw,
                           const float* __restrict__ cb, float* __restrict__ out) {
    int i = threadIdx.x;
    if (i >= B_SZ * 4) return;
    int b = i >> 2, k = i & 3;
    const float* mb = mean + b * D_MODEL;
    const float* wk = cw + k * D_MODEL;
    float s = cb[k];
    for (int d = 0; d < D_MODEL; ++d) s = fmaf(mb[d], wk[d], s);
    out[b * 4 + k] = s;
}

// ---------------- host launcher ----------------
extern "C" void kernel_launch(void* const* d_in, const int* in_sizes, int n_in,
                              void* d_out, int out_size, void* d_ws, size_t ws_size,
                              hipStream_t stream) {
    const float* x_in      = (const float*)d_in[0];
    const float* in_proj_w = (const float*)d_in[1];
    const float* conv_w    = (const float*)d_in[2];
    const float* conv_b    = (const float*)d_in[3];
    const float* dt_bias   = (const float*)d_in[4];
    const float* A_log     = (const float*)d_in[5];
    const float* D_param   = (const float*)d_in[6];
    const float* gnorm_w   = (const float*)d_in[7];
    const float* out_proj  = (const float*)d_in[8];
    const float* norm_w    = (const float*)d_in[9];
    const float* cls_w     = (const float*)d_in[10];
    const float* cls_b     = (const float*)d_in[11];
    float* out = (float*)d_out;

    char* w = (char*)d_ws;
    size_t off = 0;
    auto take = [&](size_t n) { void* p = w + off; off += (n + 255) & ~(size_t)255; return p; };

    unsigned short* XH  = (unsigned short*)take((size_t)ROWS * D_MODEL * 2);
    unsigned short* XL  = (unsigned short*)take((size_t)ROWS * D_MODEL * 2);
    unsigned short* W1H = (unsigned short*)take((size_t)D_IN_PROJ * D_MODEL * 2);
    unsigned short* W1L = (unsigned short*)take((size_t)D_IN_PROJ * D_MODEL * 2);
    float* ZX    = (float*)take((size_t)ROWS * D_IN_PROJ * 4);
    float* XCONV = (float*)take((size_t)ROWS * CONV_DIM * 4);
    float* DT    = (float*)take((size_t)ROWS * NHEADS * 4);
    float* Y     = (float*)take((size_t)ROWS * D_INNER * 4);
    unsigned short* YH = (unsigned short*)take((size_t)ROWS * D_INNER * 2);
    unsigned short* YL = (unsigned short*)take((size_t)ROWS * D_INNER * 2);
    float* MEANB = (float*)take((size_t)B_SZ * D_MODEL * 4);
    // overlays (lifetimes disjoint in stream order):
    float* OUTM = (float*)XH;              // XH+XL (64MB) dead after GEMM1; OUTM = 64MB
    unsigned short* W2H = W1H;             // W1 dead after GEMM1; W2 is smaller
    unsigned short* W2L = W1L;

    // 1) stage x and in_proj_w as split-bf16
    {
        int n = ROWS * D_MODEL;
        split_bf16_kernel<<<(n + 255) / 256, 256, 0, stream>>>(x_in, XH, XL, n);
        n = D_IN_PROJ * D_MODEL;
        split_bf16_kernel<<<(n + 255) / 256, 256, 0, stream>>>(in_proj_w, W1H, W1L, n);
    }
    // 2) zxbcdt = x @ in_proj_w^T  (WMMA, 32x32 tile/wave)
    {
        dim3 grid(ROWS / 256, D_IN_PROJ / 32);   // (64, 137)
        gemm_bf16x3_kernel<<<grid, 256, 0, stream>>>(XH, XL, W1H, W1L, ZX, D_IN_PROJ, D_MODEL);
    }
    // 3) dt = softplus(dt + dt_bias)
    dt_softplus_kernel<<<(ROWS * NHEADS + 255) / 256, 256, 0, stream>>>(ZX, dt_bias, DT);
    // 4) causal depthwise conv + silu
    {
        dim3 grid((CONV_DIM + 255) / 256, ROWS);
        conv_silu_kernel<<<grid, 256, 0, stream>>>(ZX, conv_w, conv_b, XCONV);
    }
    // 5) sequential SSD scan (+ D*x skip)
    ssd_scan_kernel<<<B_SZ * NHEADS, 256, 0, stream>>>(XCONV, DT, A_log, D_param, Y);
    // 6) gate with silu(z), group RMSNorm, emit split-bf16
    gate_gnorm_kernel<<<ROWS, 256, 0, stream>>>(Y, ZX, gnorm_w, YH, YL);
    // 7) stage out_proj_w, then out = y @ out_proj_w^T (WMMA, 32x32 tile/wave)
    {
        int n = D_MODEL * D_INNER;
        split_bf16_kernel<<<(n + 255) / 256, 256, 0, stream>>>(out_proj, W2H, W2L, n);
        dim3 grid(ROWS / 256, D_MODEL / 32);     // (64, 32)
        gemm_bf16x3_kernel<<<grid, 256, 0, stream>>>(YH, YL, W2H, W2L, OUTM, D_MODEL, D_INNER);
    }
    // 8) final RMSNorm (scaled by 1/L), mean over L, classifier head
    rmsnorm_rows_kernel<<<ROWS, 256, 0, stream>>>(OUTM, norm_w);
    mean_reduce_kernel<<<(B_SZ * D_MODEL + 255) / 256, 256, 0, stream>>>(OUTM, MEANB);
    cls_kernel<<<1, 32, 0, stream>>>(MEANB, cls_w, cls_b, out);
}
